// TransXLDecoderBlock_45423574123145
// MI455X (gfx1250) — matrix operand order repaired and checked by reading, not compiled, not run on hardware
//
#include <hip/hip_runtime.h>
#include <hip/hip_bf16.h>

// Transformer-XL decoder block for gfx1250 (MI455X). All GEMM/attention math
// via v_wmma_f32_16x16x32_bf16 (f32 accumulate); flash-attention streaming
// softmax (S x K score matrix never materialized); rel-shift handled as a
// rolling 64-wide diagonal window of (q+v) @ r^T kept in LDS.

typedef __attribute__((ext_vector_type(8)))  float  f32x8;
typedef __attribute__((ext_vector_type(16))) __bf16 bf16x16;
typedef __attribute__((ext_vector_type(8)))  __bf16 bf16x8;
typedef int v4i_g __attribute__((vector_size(16)));   // for async-to-LDS builtin

#define S_LEN  2048
#define B_SZ   2
#define E_DIM  512
#define H_N    8
#define I_DIM  64
#define M_LEN  2048
#define K_LEN  4096
#define RP     4144          // K_LEN + 48: r-table pad (window reads t <= K+31)
#define FF_DIM 2048
#define ROWS   (S_LEN * B_SZ)    // 4096
#define CATROWS (K_LEN * B_SZ)   // 8192

#if defined(__gfx1250__) && __has_builtin(__builtin_amdgcn_global_load_async_to_lds_b128) && __has_builtin(__builtin_amdgcn_s_wait_asynccnt)
#define ASYNC_LDS 1
#else
#define ASYNC_LDS 0
#endif

__device__ __forceinline__ bf16x16 pack16(bf16x8 lo, bf16x8 hi) {
  return __builtin_shufflevector(lo, hi, 0,1,2,3,4,5,6,7,8,9,10,11,12,13,14,15);
}
__device__ __forceinline__ f32x8 zero8() {
  f32x8 z;
#pragma unroll
  for (int i = 0; i < 8; ++i) z[i] = 0.f;
  return z;
}

// ---------------------------------------------------------------- conversions
__global__ __launch_bounds__(256) void cvt_bf16(const float* __restrict__ src,
                                                __bf16* __restrict__ dst, int n) {
  int i = blockIdx.x * 256 + threadIdx.x;
  if (i < n) dst[i] = (__bf16)src[i];
}

__global__ __launch_bounds__(256) void build_cat(const float* __restrict__ mem,
                                                 const float* __restrict__ inp,
                                                 __bf16* __restrict__ dst) {
  int i = blockIdx.x * 256 + threadIdx.x;
  const int half = M_LEN * B_SZ * E_DIM;
  if (i < half) dst[i] = (__bf16)mem[i];
  else if (i < 2 * half) dst[i] = (__bf16)inp[i - half];
}

// ------------------------------------------------------------------ WMMA GEMM
// C = A(bf16, row-major, Mrows x Kd) @ B(bf16, row-major, Kd x N), f32 accum.
// Block tile 128x128, K-chunk 32, 8 waves (4x2), wave tile 32x64 = 2x4 WMMA.
enum { EP_QKV = 0, EP_R = 1, EP_F32 = 2, EP_RELU = 3, EP_BIAS = 4 };

template <int MODE>
__global__ __launch_bounds__(256)
void wmma_gemm(const __bf16* __restrict__ A, const __bf16* __restrict__ Bm,
               int Kd, int N, float* __restrict__ outF,
               __bf16* __restrict__ o0, __bf16* __restrict__ o1,
               __bf16* __restrict__ o2, __bf16* __restrict__ o3,
               const float* __restrict__ aux0, const float* __restrict__ aux1) {
  __shared__ __align__(16) __bf16 As[128][48];   // row stride 96 B (16B-aligned)
  __shared__ __align__(16) __bf16 Bs[128][48];   // transposed: Bs[n][k]
  const int tid = threadIdx.x;
  const int lane = tid & 31, wave = tid >> 5;
  const int l15 = lane & 15, hs = lane >> 4;
  const int wm = wave >> 1, wn = wave & 1;
  const int m0 = blockIdx.y * 128, n0 = blockIdx.x * 128;

  f32x8 acc[2][4];
#pragma unroll
  for (int a = 0; a < 2; ++a)
#pragma unroll
    for (int b = 0; b < 4; ++b) acc[a][b] = zero8();

  for (int kc = 0; kc < Kd; kc += 32) {
#pragma unroll
    for (int it = 0; it < 2; ++it) {               // A tile: 128x32 global->LDS
      int idx = tid + it * 256;
      int r = idx >> 2, cs = (idx & 3) * 8;
      const __bf16* gp = A + (size_t)(m0 + r) * Kd + kc + cs;
#if ASYNC_LDS
      __builtin_amdgcn_global_load_async_to_lds_b128(
          (__attribute__((address_space(1))) v4i_g*)gp,
          (__attribute__((address_space(3))) v4i_g*)&As[r][cs], 0, 0);
#else
      *(bf16x8*)&As[r][cs] = *(const bf16x8*)gp;
#endif
    }
#pragma unroll
    for (int it = 0; it < 2; ++it) {               // B tile: 32x128, transpose
      int idx = tid + it * 256;
      int kk = idx >> 4, ns = (idx & 15) * 8;
      bf16x8 bv = *(const bf16x8*)(Bm + (size_t)(kc + kk) * N + n0 + ns);
#pragma unroll
      for (int q = 0; q < 8; ++q) Bs[ns + q][kk] = bv[q];
    }
#if ASYNC_LDS
    __builtin_amdgcn_s_wait_asynccnt(0);
#endif
    __syncthreads();

    bf16x16 av[2], bv[4];
#pragma unroll
    for (int am = 0; am < 2; ++am) {
      int row = wm * 32 + am * 16 + l15;           // A: lanes<16 K0-7/16-23, >=16 K8-15/24-31
      av[am] = pack16(*(const bf16x8*)&As[row][hs * 8],
                      *(const bf16x8*)&As[row][16 + hs * 8]);
    }
#pragma unroll
    for (int bn = 0; bn < 4; ++bn) {
      int col = wn * 64 + bn * 16 + l15;           // B: lane = column, K0-7/16-23 | K8-15/24-31
      bv[bn] = pack16(*(const bf16x8*)&Bs[col][hs * 8],
                      *(const bf16x8*)&Bs[col][16 + hs * 8]);
    }
#pragma unroll
    for (int am = 0; am < 2; ++am)
#pragma unroll
      for (int bn = 0; bn < 4; ++bn)
        acc[am][bn] = __builtin_amdgcn_wmma_f32_16x16x32_bf16(
            false, av[am], false, bv[bn], (short)0, acc[am][bn], false, false);
    __syncthreads();
  }

#pragma unroll
  for (int am = 0; am < 2; ++am)
#pragma unroll
    for (int bn = 0; bn < 4; ++bn)
#pragma unroll
      for (int vv = 0; vv < 8; ++vv) {
        int row = m0 + wm * 32 + am * 16 + hs * 8 + vv;
        int col = n0 + wn * 64 + bn * 16 + l15;
        float val = acc[am][bn][vv];
        if constexpr (MODE == EP_QKV) {
          int k = row >> 1, b = row & 1;           // row = k*B + b
          int sec = col >> 9, cc = col & 511;
          int h = cc >> 6, i = cc & 63;
          if (sec == 0) {                          // Q: keep last S rows, add u/v
            if (k >= M_LEN) {
              size_t qi = (((size_t)(b * H_N + h)) * S_LEN + (k - M_LEN)) * I_DIM + i;
              o0[qi] = (__bf16)(val + aux0[cc]);   // q + u
              o1[qi] = (__bf16)(val + aux1[cc]);   // q + v
            }
          } else if (sec == 1) {                   // K: [bh][j][i]
            o2[(((size_t)(b * H_N + h)) * K_LEN + k) * I_DIM + i] = (__bf16)val;
          } else {                                 // V transposed: [bh][i][j]
            o3[(((size_t)(b * H_N + h)) * I_DIM + i) * K_LEN + k] = (__bf16)val;
          }
        } else if constexpr (MODE == EP_R) {       // r: [h][t][i]
          o0[(((size_t)(col >> 6)) * RP + row) * I_DIM + (col & 63)] = (__bf16)val;
        } else if constexpr (MODE == EP_F32) {
          outF[(size_t)row * N + col] = val;
        } else if constexpr (MODE == EP_RELU) {
          o0[(size_t)row * N + col] = (__bf16)fmaxf(val + aux0[col], 0.f);
        } else {                                   // EP_BIAS
          outF[(size_t)row * N + col] = val + aux0[col];
        }
      }
}

// ----------------------------------------------------- flash attention (TXL)
// One wave = 32 query rows (2 row-tiles) for one (b,h); streams 32 keys/iter.
// BD[s,j] = (q_s+v) . r[(j-s)+S-1]: the needed r-window per (32q x 32k) step
// is 64 wide and slides by 32 -> rolling 64-col LDS buffer, only 2 new 16-col
// D2 tiles computed per iteration. K/r/V B-operands are shared by both query
// row-tiles (halves L2 operand traffic per WMMA).
__global__ __launch_bounds__(128)
void attn_fa(const __bf16* __restrict__ quB, const __bf16* __restrict__ qvB,
             const __bf16* __restrict__ kB, const __bf16* __restrict__ vTB,
             const __bf16* __restrict__ rB, __bf16* __restrict__ attnBF) {
  __shared__ float d2buf[4][32][64];               // rolling diagonal window
  __shared__ __align__(16) __bf16 pbuf[4][32][48];
  const int tid = threadIdx.x;
  const int wave = tid >> 5, lane = tid & 31;
  const int l15 = lane & 15, hs = lane >> 4;
  const int bh = blockIdx.x, h = bh & 7, b = bh >> 3;
  const int s0 = blockIdx.y * 128 + wave * 32;

  bf16x16 qua[2][2], qva[2][2];                    // [row-tile][i-chunk]
#pragma unroll
  for (int qt = 0; qt < 2; ++qt) {
    const __bf16* qr = quB + ((size_t)bh * S_LEN + s0 + qt * 16 + l15) * I_DIM;
    const __bf16* vr = qvB + ((size_t)bh * S_LEN + s0 + qt * 16 + l15) * I_DIM;
#pragma unroll
    for (int c = 0; c < 2; ++c) {
      qua[qt][c] = pack16(*(const bf16x8*)(qr + c * 32 + hs * 8),
                          *(const bf16x8*)(qr + c * 32 + 16 + hs * 8));
      qva[qt][c] = pack16(*(const bf16x8*)(vr + c * 32 + hs * 8),
                          *(const bf16x8*)(vr + c * 32 + 16 + hs * 8));
    }
  }

  f32x8 acc[2][4];
#pragma unroll
  for (int qt = 0; qt < 2; ++qt)
#pragma unroll
    for (int t = 0; t < 4; ++t) acc[qt][t] = zero8();
  float mrow[2][8], lrow[2][8];
#pragma unroll
  for (int qt = 0; qt < 2; ++qt)
#pragma unroll
    for (int v = 0; v < 8; ++v) { mrow[qt][v] = -3.0e38f; lrow[qt][v] = 0.f; }

  const __bf16* kbase = kB + (size_t)bh * K_LEN * I_DIM;
  const __bf16* vbase = vTB + (size_t)bh * I_DIM * K_LEN;
  const __bf16* rbase = rB + (size_t)h * RP * I_DIM;
  const int jend = s0 + M_LEN + 32;                // exclusive; multiple of 32

  // One D2 window tile: 16 t-cols at absolute base T, both query row-tiles.
  auto d2_tile = [&](int T) {
    bf16x16 rb[2];
#pragma unroll
    for (int c = 0; c < 2; ++c) {
      const __bf16* rp = rbase + (size_t)(T + l15) * I_DIM + c * 32 + hs * 8;
      rb[c] = pack16(*(const bf16x8*)rp, *(const bf16x8*)(rp + 16));
    }
#pragma unroll
    for (int qt = 0; qt < 2; ++qt) {
      f32x8 d2 = __builtin_amdgcn_wmma_f32_16x16x32_bf16(
          false, qva[qt][0], false, rb[0], (short)0, zero8(), false, false);
      d2 = __builtin_amdgcn_wmma_f32_16x16x32_bf16(
          false, qva[qt][1], false, rb[1], (short)0, d2, false, false);
#pragma unroll
      for (int v = 0; v < 8; ++v)
        d2buf[wave][qt * 16 + hs * 8 + v][((T & 63) + l15)] = d2[v];
    }
  };

  for (int j0 = 0; j0 < jend; j0 += 32) {
    const int tb = j0 - s0 + (S_LEN - 32);         // window covers [tb, tb+63]
    if (j0 == 0) { d2_tile(tb); d2_tile(tb + 16); }
    d2_tile(tb + 32); d2_tile(tb + 48);            // 2 new tiles, 2 carried

    bf16x16 kb[2][2];                              // [j-tile][i-chunk]
#pragma unroll
    for (int t = 0; t < 2; ++t)
#pragma unroll
      for (int c = 0; c < 2; ++c) {
        const __bf16* kp = kbase + (size_t)(j0 + t * 16 + l15) * I_DIM + c * 32 + hs * 8;
        kb[t][c] = pack16(*(const bf16x8*)kp, *(const bf16x8*)(kp + 16));
      }
    f32x8 ac[2][2];                                // [row-tile][j-tile]
#pragma unroll
    for (int qt = 0; qt < 2; ++qt)
#pragma unroll
      for (int t = 0; t < 2; ++t) {
        ac[qt][t] = __builtin_amdgcn_wmma_f32_16x16x32_bf16(
            false, qua[qt][0], false, kb[t][0], (short)0, zero8(), false, false);
        ac[qt][t] = __builtin_amdgcn_wmma_f32_16x16x32_bf16(
            false, qua[qt][1], false, kb[t][1], (short)0, ac[qt][t], false, false);
      }
    __builtin_amdgcn_wave_barrier();               // same-wave LDS is in-order;
    __asm__ __volatile__("" ::: "memory");         // block compiler reordering

#pragma unroll
    for (int qt = 0; qt < 2; ++qt) {
#pragma unroll
      for (int v = 0; v < 8; ++v) {
        int rr = qt * 16 + hs * 8 + v;
        float bd0 = d2buf[wave][rr][(tb + l15 - rr + 31) & 63];
        float bd1 = d2buf[wave][rr][(tb + 16 + l15 - rr + 31) & 63];
        float sc0 = (ac[qt][0][v] + bd0) * 0.125f; // 1/sqrt(I)
        float sc1 = (ac[qt][1][v] + bd1) * 0.125f;
        sc0 = (j0 + l15 <= s0 + rr + M_LEN) ? sc0 : -1.0e30f;
        sc1 = (j0 + 16 + l15 <= s0 + rr + M_LEN) ? sc1 : -1.0e30f;
        float tmax = fmaxf(sc0, sc1);
#pragma unroll
        for (int off = 1; off < 16; off <<= 1) tmax = fmaxf(tmax, __shfl_xor(tmax, off, 32));
        float nm = fmaxf(mrow[qt][v], tmax);
        float e0 = __expf(sc0 - nm), e1 = __expf(sc1 - nm);
        float cr = __expf(mrow[qt][v] - nm);
        float rs = e0 + e1;
#pragma unroll
        for (int off = 1; off < 16; off <<= 1) rs += __shfl_xor(rs, off, 32);
        lrow[qt][v] = lrow[qt][v] * cr + rs;
        mrow[qt][v] = nm;
#pragma unroll
        for (int t = 0; t < 4; ++t) acc[qt][t][v] *= cr;
        pbuf[wave][rr][l15] = (__bf16)e0;
        pbuf[wave][rr][16 + l15] = (__bf16)e1;
      }
    }
    __builtin_amdgcn_wave_barrier();
    __asm__ __volatile__("" ::: "memory");
    bf16x16 pa[2];
#pragma unroll
    for (int qt = 0; qt < 2; ++qt)
      pa[qt] = pack16(*(const bf16x8*)&pbuf[wave][qt * 16 + l15][hs * 8],
                      *(const bf16x8*)&pbuf[wave][qt * 16 + l15][16 + hs * 8]);
#pragma unroll
    for (int t4 = 0; t4 < 4; ++t4) {               // acc += P @ V (V^T layout)
      const __bf16* vp = vbase + (size_t)(t4 * 16 + l15) * K_LEN + j0 + hs * 8;
      bf16x16 vb = pack16(*(const bf16x8*)vp, *(const bf16x8*)(vp + 16));
#pragma unroll
      for (int qt = 0; qt < 2; ++qt)
        acc[qt][t4] = __builtin_amdgcn_wmma_f32_16x16x32_bf16(
            false, pa[qt], false, vb, (short)0, acc[qt][t4], false, false);
    }
  }

#pragma unroll
  for (int qt = 0; qt < 2; ++qt)
#pragma unroll
    for (int t4 = 0; t4 < 4; ++t4)
#pragma unroll
      for (int v = 0; v < 8; ++v) {
        int s = s0 + qt * 16 + hs * 8 + v;
        int i = t4 * 16 + l15;
        attnBF[((size_t)s * B_SZ + b) * (H_N * I_DIM) + h * I_DIM + i] =
            (__bf16)(acc[qt][t4][v] / lrow[qt][v]);
      }
}

// ------------------------------------------------------- residual + layernorm
template <int WRITE_BF>
__global__ __launch_bounds__(256)
void ln_fuse(const float* __restrict__ xa, const float* __restrict__ xb,
             const float* __restrict__ g, const float* __restrict__ bt,
             float* __restrict__ outF, __bf16* __restrict__ outBF) {
  __shared__ float red[16];
  const int row = blockIdx.x, tid = threadIdx.x;
  const size_t base = (size_t)row * E_DIM;
  float x0 = xa[base + tid] + xb[base + tid];
  float x1 = xa[base + tid + 256] + xb[base + tid + 256];
  float s = x0 + x1, sq = x0 * x0 + x1 * x1;
#pragma unroll
  for (int off = 1; off < 32; off <<= 1) {
    s += __shfl_xor(s, off, 32);
    sq += __shfl_xor(sq, off, 32);
  }
  const int wave = tid >> 5;
  if ((tid & 31) == 0) { red[wave] = s; red[8 + wave] = sq; }
  __syncthreads();
  float ts = 0.f, tq = 0.f;
#pragma unroll
  for (int wv = 0; wv < 8; ++wv) { ts += red[wv]; tq += red[8 + wv]; }
  float mu = ts * (1.0f / E_DIM);
  float var = tq * (1.0f / E_DIM) - mu * mu;
  float rinv = rsqrtf(var + 1e-5f);
  float y0 = (x0 - mu) * rinv * g[tid] + bt[tid];
  float y1 = (x1 - mu) * rinv * g[tid + 256] + bt[tid + 256];
  outF[base + tid] = y0;
  outF[base + tid + 256] = y1;
  if constexpr (WRITE_BF) {
    outBF[base + tid] = (__bf16)y0;
    outBF[base + tid + 256] = (__bf16)y1;
  }
}

// --------------------------------------------------------------------- driver
extern "C" void kernel_launch(void* const* d_in, const int* in_sizes, int n_in,
                              void* d_out, int out_size, void* d_ws, size_t ws_size,
                              hipStream_t stream) {
  (void)in_sizes; (void)n_in; (void)out_size; (void)ws_size;
  const float* inputDec = (const float*)d_in[0];
  const float* posEmb   = (const float*)d_in[1];
  const float* u_p      = (const float*)d_in[2];
  const float* v_p      = (const float*)d_in[3];
  const float* memories = (const float*)d_in[4];
  const float* Wqkv     = (const float*)d_in[5];
  const float* Wr       = (const float*)d_in[6];
  const float* Wo       = (const float*)d_in[7];
  const float* ln1g     = (const float*)d_in[8];
  const float* ln1b     = (const float*)d_in[9];
  const float* W1       = (const float*)d_in[10];
  const float* b1       = (const float*)d_in[11];
  const float* W2       = (const float*)d_in[12];
  const float* b2       = (const float*)d_in[13];
  const float* ln2g     = (const float*)d_in[14];
  const float* ln2b     = (const float*)d_in[15];
  float* out = (float*)d_out;

  char* w = (char*)d_ws;
  size_t off = 0;
  auto alloc = [&](size_t bytes) {
    void* p = w + off;
    off = (off + bytes + 255) & ~(size_t)255;
    return p;
  };

  __bf16* catBF  = (__bf16*)alloc((size_t)CATROWS * E_DIM * 2);
  __bf16* WqkvBF = (__bf16*)alloc((size_t)E_DIM * 3 * H_N * I_DIM * 2);
  __bf16* posBF  = (__bf16*)alloc((size_t)K_LEN * E_DIM * 2);
  __bf16* WrBF   = (__bf16*)alloc((size_t)E_DIM * E_DIM * 2);
  __bf16* WoBF   = (__bf16*)alloc((size_t)E_DIM * E_DIM * 2);
  __bf16* W1BF   = (__bf16*)alloc((size_t)E_DIM * FF_DIM * 2);
  __bf16* W2BF   = (__bf16*)alloc((size_t)FF_DIM * E_DIM * 2);
  __bf16* quBF   = (__bf16*)alloc((size_t)B_SZ * H_N * S_LEN * I_DIM * 2);
  __bf16* qvBF   = (__bf16*)alloc((size_t)B_SZ * H_N * S_LEN * I_DIM * 2);
  __bf16* kBF    = (__bf16*)alloc((size_t)B_SZ * H_N * K_LEN * I_DIM * 2);
  __bf16* vTBF   = (__bf16*)alloc((size_t)B_SZ * H_N * I_DIM * K_LEN * 2);
  __bf16* rBF    = (__bf16*)alloc((size_t)H_N * RP * I_DIM * 2);
  __bf16* attnBF = (__bf16*)alloc((size_t)ROWS * E_DIM * 2);
  float*  attnWo = (float*)alloc((size_t)ROWS * E_DIM * 4);
  float*  xF     = (float*)alloc((size_t)ROWS * E_DIM * 4);
  __bf16* xBF    = (__bf16*)alloc((size_t)ROWS * E_DIM * 2);
  __bf16* ffBF   = (__bf16*)alloc((size_t)ROWS * FF_DIM * 2);
  float*  ff2    = (float*)alloc((size_t)ROWS * E_DIM * 4);

  int n;
  n = CATROWS * E_DIM;
  build_cat<<<n / 256, 256, 0, stream>>>(memories, inputDec, catBF);
  n = E_DIM * 3 * H_N * I_DIM;
  cvt_bf16<<<n / 256, 256, 0, stream>>>(Wqkv, WqkvBF, n);
  n = K_LEN * E_DIM;
  cvt_bf16<<<n / 256, 256, 0, stream>>>(posEmb, posBF, n);
  n = E_DIM * E_DIM;
  cvt_bf16<<<n / 256, 256, 0, stream>>>(Wr, WrBF, n);
  cvt_bf16<<<n / 256, 256, 0, stream>>>(Wo, WoBF, n);
  n = E_DIM * FF_DIM;
  cvt_bf16<<<n / 256, 256, 0, stream>>>(W1, W1BF, n);
  cvt_bf16<<<n / 256, 256, 0, stream>>>(W2, W2BF, n);

  // qkv = cat @ Wqkv, scatter-epilogue into (q+u), (q+v), K, V^T
  wmma_gemm<EP_QKV><<<dim3(12, 64), 256, 0, stream>>>(
      catBF, WqkvBF, E_DIM, 3 * H_N * I_DIM, nullptr,
      quBF, qvBF, kBF, vTBF, u_p, v_p);
  // r = pos @ Wr
  wmma_gemm<EP_R><<<dim3(4, 32), 256, 0, stream>>>(
      posBF, WrBF, E_DIM, E_DIM, nullptr,
      rBF, nullptr, nullptr, nullptr, nullptr, nullptr);
  // streaming attention (AC + rel-shifted BD, online softmax, P@V)
  attn_fa<<<dim3(B_SZ * H_N, S_LEN / 128), 128, 0, stream>>>(
      quBF, qvBF, kBF, vTBF, rBF, attnBF);
  // attn @ Wo
  wmma_gemm<EP_F32><<<dim3(4, 32), 256, 0, stream>>>(
      attnBF, WoBF, E_DIM, E_DIM, attnWo,
      nullptr, nullptr, nullptr, nullptr, nullptr, nullptr);
  // LN1(inputDec + attn@Wo) -> xF (f32) + xBF (bf16)
  ln_fuse<1><<<ROWS, 256, 0, stream>>>(inputDec, attnWo, ln1g, ln1b, xF, xBF);
  // relu(x @ W1 + b1) -> ffBF
  wmma_gemm<EP_RELU><<<dim3(16, 32), 256, 0, stream>>>(
      xBF, W1BF, E_DIM, FF_DIM, nullptr,
      ffBF, nullptr, nullptr, nullptr, b1, nullptr);
  // ff @ W2 + b2 -> ff2
  wmma_gemm<EP_BIAS><<<dim3(4, 32), 256, 0, stream>>>(
      ffBF, W2BF, FF_DIM, E_DIM, ff2,
      nullptr, nullptr, nullptr, nullptr, b2, nullptr);
  // LN2(x + ff) -> out
  ln_fuse<0><<<ROWS, 256, 0, stream>>>(xF, ff2, ln2g, ln2b, out, nullptr);
}
